// KBLamLlamaAttention_7155415515455
// MI455X (gfx1250) — matrix operand order repaired and checked by reading, not compiled, not run on hardware
//
#include <hip/hip_runtime.h>

typedef __bf16 bf16;
typedef __attribute__((ext_vector_type(16))) __bf16 v16bf;
typedef __attribute__((ext_vector_type(8)))  __bf16 v8bf;
typedef __attribute__((ext_vector_type(8)))  float   v8f;

constexpr int Bc = 2, Sc = 1024, HIDc = 2048, Hc = 32, KVc = 8, HDc = 64;
constexpr int GROUPSc = Hc / KVc;
constexpr int KBN = 4096, TOPK = 128;
constexpr float SCALE = 0.125f;             // 1/sqrt(64)
constexpr float NEG   = -1000000000.0f;     // reference mask value

static __device__ __forceinline__ v16bf make_frag(v8bf lo, v8bf hi) {
  v16bf r;
#pragma unroll
  for (int i = 0; i < 8; ++i) { r[i] = lo[i]; r[8 + i] = hi[i]; }
  return r;
}

// A operand (16 x 32 bf16): lane m = l&15; lanes 0-15 hold K 0-7 & 16-23, lanes 16-31 hold K 8-15 & 24-31.
static __device__ __forceinline__ v16bf load_a_frag(const bf16* base, int ldr) {
  const int l  = threadIdx.x & 31;
  const int m  = l & 15;
  const int h2 = l >> 4;
  const bf16* p = base + m * ldr + h2 * 8;
  v8bf lo = *(const v8bf*)(p);
  v8bf hi = *(const v8bf*)(p + 16);
  return make_frag(lo, hi);
}

// B operand (32 x 16 bf16): lane n = l&15; lanes 0-15 hold K 0-7 & 16-23, lanes 16-31 hold K 8-15 & 24-31.
static __device__ __forceinline__ v16bf load_b_frag(const bf16* base, int ldr) {
  const int l = threadIdx.x & 31;
  const bf16* p = base + (l & 15) + (l >> 4) * 8 * ldr;
  v16bf r;
#pragma unroll
  for (int i = 0; i < 8; ++i) { r[i] = p[i * ldr]; r[8 + i] = p[(16 + i) * ldr]; }
  return r;
}

static __device__ __forceinline__ v8f wmma_bf16(v16bf a, v16bf b, v8f c) {
  return __builtin_amdgcn_wmma_f32_16x16x32_bf16(false, a, false, b, (short)0, c, false, false);
}

// gfx1250 async global->LDS copy (GVS mode): 16B per lane, ASYNCcnt tracked.
// LDS byte address = low 32 bits of the flat pointer to the __shared__ object.
static __device__ __forceinline__ void async_ld_b128(void* lds_dst, const void* gbase, unsigned voff_bytes) {
  unsigned lds_addr = (unsigned)(unsigned long long)lds_dst;
  asm volatile("global_load_async_to_lds_b128 %0, %1, %2 offset:0"
               :: "v"(lds_addr), "v"(voff_bytes), "s"((unsigned long long)gbase)
               : "memory");
}
static __device__ __forceinline__ void async_wait0() {
  asm volatile("s_wait_asynccnt 0x0" ::: "memory");
}

// ---------------------------------------------------------------------------
// Generic C[M,N] = A[M,K] * W[N,K]^T, bf16 WMMA, f32 accumulate/output.
// Block: 256 threads (8 waves), 128x128 tile; wave tile 32x64; k-step 64.
// ---------------------------------------------------------------------------
template <bool A_BF16>
__global__ __launch_bounds__(256) void gemm_xwt_kernel(const void* __restrict__ Aptr,
                                                       const float* __restrict__ W,
                                                       float* __restrict__ C,
                                                       int M, int N, int K) {
  __shared__ __align__(16) bf16 As[128][72];    // rows m, cols k (pad 72 -> conflict-free A frags)
  __shared__ __align__(16) bf16 Bs[64][136];    // rows k, cols n (transposed weight tile)
  const int m0  = blockIdx.y * 128;
  const int n0  = blockIdx.x * 128;
  const int tid = threadIdx.x;
  const int w   = tid >> 5;
  const int my  = (w >> 1) * 32;
  const int nx  = (w & 1) * 64;
  const int row  = tid >> 1;
  const int koff = (tid & 1) * 32;

  const v8f zero = {0.f,0.f,0.f,0.f,0.f,0.f,0.f,0.f};
  v8f acc[2][4];
#pragma unroll
  for (int i = 0; i < 2; ++i)
#pragma unroll
    for (int j = 0; j < 4; ++j) acc[i][j] = zero;

  for (int kt = 0; kt < K; kt += 64) {
    __syncthreads();
    if constexpr (A_BF16) {
#pragma unroll
      for (int i = 0; i < 4; ++i)
        async_ld_b128(&As[row][koff + i * 8], (const bf16*)Aptr + (size_t)m0 * K,
                      (unsigned)(((unsigned)row * K + kt + koff + i * 8) * 2));
    } else {
      const float* gA = (const float*)Aptr + (size_t)(m0 + row) * K + kt + koff;
      if (kt + 64 < K) __builtin_prefetch(gA + 64, 0, 1);
#pragma unroll
      for (int i = 0; i < 8; ++i) {
        float4 f = *(const float4*)(gA + i * 4);
        As[row][koff + i*4 + 0] = (bf16)f.x;
        As[row][koff + i*4 + 1] = (bf16)f.y;
        As[row][koff + i*4 + 2] = (bf16)f.z;
        As[row][koff + i*4 + 3] = (bf16)f.w;
      }
    }
    {
      const float* gW = W + (size_t)(n0 + row) * K + kt + koff;
      if (kt + 64 < K) __builtin_prefetch(gW + 64, 0, 1);
#pragma unroll
      for (int i = 0; i < 8; ++i) {
        float4 f = *(const float4*)(gW + i * 4);
        Bs[koff + i*4 + 0][row] = (bf16)f.x;
        Bs[koff + i*4 + 1][row] = (bf16)f.y;
        Bs[koff + i*4 + 2][row] = (bf16)f.z;
        Bs[koff + i*4 + 3][row] = (bf16)f.w;
      }
    }
    if constexpr (A_BF16) async_wait0();
    __syncthreads();

    v16bf a00 = load_a_frag(&As[my][0],       72);
    v16bf a01 = load_a_frag(&As[my][32],      72);
    v16bf a10 = load_a_frag(&As[my + 16][0],  72);
    v16bf a11 = load_a_frag(&As[my + 16][32], 72);
#pragma unroll
    for (int j = 0; j < 4; ++j) {
      v16bf b0 = load_b_frag(&Bs[0][nx + j * 16],  136);
      v16bf b1 = load_b_frag(&Bs[32][nx + j * 16], 136);
      acc[0][j] = wmma_bf16(a00, b0, acc[0][j]);
      acc[0][j] = wmma_bf16(a01, b1, acc[0][j]);
      acc[1][j] = wmma_bf16(a10, b0, acc[1][j]);
      acc[1][j] = wmma_bf16(a11, b1, acc[1][j]);
    }
  }

  const int l = tid & 31, lh = l >> 4, ln = l & 15;
#pragma unroll
  for (int mi = 0; mi < 2; ++mi)
#pragma unroll
    for (int j = 0; j < 4; ++j)
#pragma unroll
      for (int r = 0; r < 8; ++r) {
        const int rr = m0 + my + mi * 16 + r + 8 * lh;
        const int cc = n0 + nx + j * 16 + ln;
        C[(size_t)rr * N + cc] = acc[mi][j][r];
      }
}

// ---------------------------------------------------------------------------
// RoPE + (b,s,h,d)->(b,h,s,d) relayout + f32->bf16.
// ---------------------------------------------------------------------------
__global__ void rope_relayout_kernel(const float* __restrict__ src, bf16* __restrict__ dst,
                                     int nheads, int do_rope) {
  const int idx = blockIdx.x * blockDim.x + threadIdx.x;
  const int d = idx & 63;
  const int s = (idx >> 6) & (Sc - 1);
  const int t = idx >> 16;               // S*HD == 65536
  const int hh = t % nheads;
  const int b  = t / nheads;
  const size_t srcbase = ((size_t)b * Sc + s) * ((size_t)nheads * HDc) + (size_t)hh * HDc;
  const float x = src[srcbase + d];
  float out = x;
  if (do_rope) {
    const int i = d & 31;
    const float inv = __powf(10000.0f, -(float)i * (1.0f / 32.0f));
    float sn, cs;
    __sincosf((float)s * inv, &sn, &cs);
    const float xr = (d < 32) ? -src[srcbase + d + 32] : src[srcbase + d - 32];
    out = x * cs + xr * sn;
  }
  dst[idx] = (bf16)out;
}

// Sum of qkb over (h,s) per (b,d): feeds the linear-algebra-reduced top-k.
__global__ __launch_bounds__(256) void sumqkb_kernel(const bf16* __restrict__ qkb,
                                                     float* __restrict__ qsum) {
  __shared__ float red[256];
  const int b = blockIdx.x >> 6;
  const int d = blockIdx.x & 63;
  float acc = 0.f;
  for (int j = threadIdx.x; j < Hc * Sc; j += 256)
    acc += (float)qkb[((size_t)b * Hc * Sc + j) * 64 + d];
  red[threadIdx.x] = acc;
  __syncthreads();
  for (int off = 128; off > 0; off >>= 1) {
    if (threadIdx.x < off) red[threadIdx.x] += red[threadIdx.x + off];
    __syncthreads();
  }
  if (threadIdx.x == 0) qsum[b * 64 + d] = red[0];
}

// Per-batch top-128 of qsum . kb_keys^T  (descending, ties -> lowest index, as jax.lax.top_k).
__global__ __launch_bounds__(256) void topk_kernel(const float* __restrict__ qsum,
                                                   const float* __restrict__ kb_keys,
                                                   int* __restrict__ top_idx) {
  __shared__ float sv[64];
  __shared__ float sc[KBN];
  __shared__ float redv[256];
  __shared__ int   redi[256];
  const int b = blockIdx.x, tid = threadIdx.x;
  if (tid < 64) sv[tid] = qsum[b * 64 + tid];
  __syncthreads();
  for (int i = 0; i < KBN / 256; ++i) {
    const int n = tid + i * 256;
    const float* kp = kb_keys + ((size_t)b * KBN + n) * 64;
    float dot = 0.f;
#pragma unroll 8
    for (int d = 0; d < 64; ++d) dot += sv[d] * kp[d];
    sc[n] = dot * SCALE;
  }
  __syncthreads();
  for (int it = 0; it < TOPK; ++it) {
    float bv = -3.0e38f; int bn = 0x7fffffff;
    for (int i = 0; i < KBN / 256; ++i) {
      const int n = tid + i * 256;
      const float v = sc[n];
      if (v > bv || (v == bv && n < bn)) { bv = v; bn = n; }
    }
    redv[tid] = bv; redi[tid] = bn;
    __syncthreads();
    for (int off = 128; off > 0; off >>= 1) {
      if (tid < off) {
        const float v2 = redv[tid + off]; const int n2 = redi[tid + off];
        if (v2 > redv[tid] || (v2 == redv[tid] && n2 < redi[tid])) { redv[tid] = v2; redi[tid] = n2; }
      }
      __syncthreads();
    }
    if (tid == 0) { top_idx[b * TOPK + it] = redi[0]; sc[redi[0]] = -3.0e38f; }
    __syncthreads();
  }
}

// kbk = bf16(kb_keys[:, :128]);  kbv = bf16(kb_values gathered at top_idx).
__global__ void gather_kernel(const float* __restrict__ kb_keys, const float* __restrict__ kb_values,
                              const int* __restrict__ top_idx, bf16* __restrict__ kbk,
                              bf16* __restrict__ kbv) {
  const int idx = blockIdx.x * blockDim.x + threadIdx.x;
  const int d = idx & 63;
  const int k = (idx >> 6) & (TOPK - 1);
  const int b = idx >> 13;
  kbk[idx] = (bf16)kb_keys[((size_t)b * KBN + k) * 64 + d];
  const int gi = top_idx[b * TOPK + k];
  kbv[idx] = (bf16)kb_values[((size_t)b * KBN + gi) * 64 + d];
}

// ---------------------------------------------------------------------------
// Fused flash-style attention with KB extension.
// Block = 128 threads (4 waves), 64 query rows; grid = B*H*(S/64).
// Causal K/V blocks first, then 2 blocks of KB columns using qkb + score_shift.
// Q/Qkb and V tiles staged with gfx1250 async global->LDS copies.
// ---------------------------------------------------------------------------
__global__ __launch_bounds__(128) void attn_kernel(const bf16* __restrict__ qbf,
                                                   const bf16* __restrict__ qkbbf,
                                                   const bf16* __restrict__ kbf,
                                                   const bf16* __restrict__ vbf,
                                                   const bf16* __restrict__ kbk,
                                                   const bf16* __restrict__ kbv,
                                                   const float* __restrict__ score_shift,
                                                   bf16* __restrict__ aout) {
  __shared__ __align__(16) bf16 Qs[64][72];
  __shared__ __align__(16) bf16 Qk[64][72];
  __shared__ __align__(16) bf16 Kt[64][72];      // transposed: [d][key]
  __shared__ __align__(16) bf16 Vs[64][72];      // [key][d]
  __shared__ __align__(16) bf16 Ps[4][16][72];   // per-wave P tile (C-layout -> A-layout bridge)

  const int qb = blockIdx.x & 15;
  const int h  = (blockIdx.x >> 4) & 31;
  const int b  = blockIdx.x >> 9;
  const int s0 = qb * 64;
  const int kvh = h / GROUPSc;
  const int tid = threadIdx.x;
  const int w = tid >> 5;
  const int l = tid & 31, lh = l >> 4, ln = l & 15;
  const float shiftv = score_shift[h];

  {
    const int r = tid >> 1, doff = (tid & 1) * 32;
    const bf16* qb_base  = qbf   + (((size_t)b * Hc + h) * Sc + s0) * 64;
    const bf16* qkb_base = qkbbf + (((size_t)b * Hc + h) * Sc + s0) * 64;
#pragma unroll
    for (int i = 0; i < 4; ++i) {
      const unsigned voff = (unsigned)(((unsigned)r * 64 + doff + i * 8) * 2);
      async_ld_b128(&Qs[r][doff + i * 8], qb_base,  voff);
      async_ld_b128(&Qk[r][doff + i * 8], qkb_base, voff);
    }
    async_wait0();
  }

  const v8f zero = {0.f,0.f,0.f,0.f,0.f,0.f,0.f,0.f};
  v8f o[4];
  float mrow[8], lrow[8];
#pragma unroll
  for (int j = 0; j < 4; ++j) o[j] = zero;
#pragma unroll
  for (int r = 0; r < 8; ++r) { mrow[r] = -3.0e38f; lrow[r] = 0.f; }

  for (int bi = 0; bi <= qb + 2; ++bi) {
    const bool is_kb = (bi > qb);
    __syncthreads();
    {
      const int key = tid >> 1, doff = (tid & 1) * 32;
      const bf16* gkb;                            // uniform K base for this block
      const bf16* gvb;                            // uniform V base for this block
      if (!is_kb) {
        const size_t base = (((size_t)b * KVc + kvh) * Sc + bi * 64) * 64;
        gkb = kbf + base; gvb = vbf + base;
      } else {
        const size_t base = ((size_t)b * TOPK + (bi - qb - 1) * 64) * 64;
        gkb = kbk + base; gvb = kbv + base;
      }
      const bf16* gk = gkb + (size_t)key * 64 + doff;
#pragma unroll
      for (int i = 0; i < 4; ++i) {
        // V: async copy straight into LDS, no VGPR round trip.
        async_ld_b128(&Vs[key][doff + i * 8], gvb,
                      (unsigned)(((unsigned)key * 64 + doff + i * 8) * 2));
        // K: needs a transpose on the LDS store, so go through VGPRs.
        v8bf kv8 = *(const v8bf*)(gk + i * 8);
#pragma unroll
        for (int e = 0; e < 8; ++e) Kt[doff + i * 8 + e][key] = kv8[e];
      }
      async_wait0();
    }
    __syncthreads();

    const bf16* qbase = is_kb ? &Qk[w * 16][0] : &Qs[w * 16][0];
    v16bf aq0 = load_a_frag(qbase, 72);        // d 0..31
    v16bf aq1 = load_a_frag(qbase + 32, 72);   // d 32..63
    v8f st[4];
#pragma unroll
    for (int j = 0; j < 4; ++j) {
      v16bf b0 = load_b_frag(&Kt[0][j * 16], 72);
      v16bf b1 = load_b_frag(&Kt[32][j * 16], 72);
      v8f s = zero;
      s = wmma_bf16(aq0, b0, s);
      s = wmma_bf16(aq1, b1, s);
      st[j] = s;
    }
#pragma unroll
    for (int j = 0; j < 4; ++j)
#pragma unroll
      for (int r = 0; r < 8; ++r) {
        float sv = st[j][r] * SCALE;
        if (is_kb) sv += shiftv;
        else if (bi == qb) {
          const int tcol = bi * 64 + j * 16 + ln;
          const int srow = s0 + w * 16 + r + 8 * lh;
          if (tcol > srow) sv += NEG;
        }
        st[j][r] = sv;
      }
    // online softmax (rows span 16 lanes of a half-wave)
#pragma unroll
    for (int r = 0; r < 8; ++r) {
      float mx = -3.0e38f;
#pragma unroll
      for (int j = 0; j < 4; ++j) mx = fmaxf(mx, st[j][r]);
      for (int dd = 8; dd > 0; dd >>= 1) mx = fmaxf(mx, __shfl_xor(mx, dd, 32));
      const float mnew = fmaxf(mrow[r], mx);
      const float fac = __expf(mrow[r] - mnew);
      mrow[r] = mnew;
      float ps = 0.f;
#pragma unroll
      for (int j = 0; j < 4; ++j) {
        const float p = __expf(st[j][r] - mnew);
        st[j][r] = p;
        ps += p;
      }
      for (int dd = 8; dd > 0; dd >>= 1) ps += __shfl_xor(ps, dd, 32);
      lrow[r] = lrow[r] * fac + ps;
#pragma unroll
      for (int j = 0; j < 4; ++j) o[j][r] *= fac;
    }
    // spill P to per-wave LDS (C layout) and reload as A operand (DS in-order per wave)
#pragma unroll
    for (int j = 0; j < 4; ++j)
#pragma unroll
      for (int r = 0; r < 8; ++r)
        Ps[w][r + 8 * lh][j * 16 + ln] = (bf16)st[j][r];
    v16bf p0 = load_a_frag(&Ps[w][0][0], 72);     // keys 0..31
    v16bf p1 = load_a_frag(&Ps[w][0][32], 72);    // keys 32..63
#pragma unroll
    for (int j = 0; j < 4; ++j) {
      v16bf vb0 = load_b_frag(&Vs[0][j * 16], 72);
      v16bf vb1 = load_b_frag(&Vs[32][j * 16], 72);
      o[j] = wmma_bf16(p0, vb0, o[j]);
      o[j] = wmma_bf16(p1, vb1, o[j]);
    }
  }

#pragma unroll
  for (int j = 0; j < 4; ++j)
#pragma unroll
    for (int r = 0; r < 8; ++r) {
      const int srow = s0 + w * 16 + r + 8 * lh;
      const float val = o[j][r] / lrow[r];
      aout[((size_t)b * Sc + srow) * HIDc + h * 64 + j * 16 + ln] = (bf16)val;
    }
}

// ---------------------------------------------------------------------------
extern "C" void kernel_launch(void* const* d_in, const int* in_sizes, int n_in,
                              void* d_out, int out_size, void* d_ws, size_t ws_size,
                              hipStream_t stream) {
  (void)in_sizes; (void)n_in; (void)out_size; (void)ws_size;
  const float* hidden     = (const float*)d_in[0];
  /* d_in[1] attention_mask: causal, reproduced analytically */
  const float* kb_keys    = (const float*)d_in[2];
  const float* kb_values  = (const float*)d_in[3];
  const float* Wq         = (const float*)d_in[4];
  const float* Wq_kb      = (const float*)d_in[5];
  const float* Wk         = (const float*)d_in[6];
  const float* Wv         = (const float*)d_in[7];
  const float* Wo         = (const float*)d_in[8];
  const float* score_shift = (const float*)d_in[9];
  /* d_in[10] position_ids == arange(S); d_in[11] top_k == 128 */
  float* out = (float*)d_out;

  char* p = (char*)d_ws;
  auto carve = [&](size_t bytes) -> char* {
    char* r = p;
    p += (bytes + 255) & ~(size_t)255;
    return r;
  };
  float* q_pre   = (float*)carve((size_t)Bc * Sc * HIDc * 4);
  float* qkb_pre = (float*)carve((size_t)Bc * Sc * HIDc * 4);
  float* k_pre   = (float*)carve((size_t)Bc * Sc * KVc * HDc * 4);
  float* v_pre   = (float*)carve((size_t)Bc * Sc * KVc * HDc * 4);
  bf16*  q_bf    = (bf16*)carve((size_t)Bc * Hc * Sc * HDc * 2);
  bf16*  qkb_bf  = (bf16*)carve((size_t)Bc * Hc * Sc * HDc * 2);
  bf16*  k_bf    = (bf16*)carve((size_t)Bc * KVc * Sc * HDc * 2);
  bf16*  v_bf    = (bf16*)carve((size_t)Bc * KVc * Sc * HDc * 2);
  bf16*  aout    = (bf16*)carve((size_t)Bc * Sc * HIDc * 2);
  float* qsum    = (float*)carve((size_t)Bc * HDc * 4);
  int*   tidx    = (int*)carve((size_t)Bc * TOPK * 4);
  bf16*  kbk     = (bf16*)carve((size_t)Bc * TOPK * HDc * 2);
  bf16*  kbv     = (bf16*)carve((size_t)Bc * TOPK * HDc * 2);

  const dim3 blk256(256), blk128(128);
  const int M = Bc * Sc;

  gemm_xwt_kernel<false><<<dim3(HIDc / 128, M / 128), blk256, 0, stream>>>(hidden, Wq,    q_pre,   M, HIDc, HIDc);
  gemm_xwt_kernel<false><<<dim3(HIDc / 128, M / 128), blk256, 0, stream>>>(hidden, Wq_kb, qkb_pre, M, HIDc, HIDc);
  gemm_xwt_kernel<false><<<dim3((KVc * HDc) / 128, M / 128), blk256, 0, stream>>>(hidden, Wk, k_pre, M, KVc * HDc, HIDc);
  gemm_xwt_kernel<false><<<dim3((KVc * HDc) / 128, M / 128), blk256, 0, stream>>>(hidden, Wv, v_pre, M, KVc * HDc, HIDc);

  rope_relayout_kernel<<<(Bc * Hc * Sc * HDc) / 256, blk256, 0, stream>>>(q_pre,   q_bf,   Hc, 1);
  rope_relayout_kernel<<<(Bc * Hc * Sc * HDc) / 256, blk256, 0, stream>>>(qkb_pre, qkb_bf, Hc, 1);
  rope_relayout_kernel<<<(Bc * KVc * Sc * HDc) / 256, blk256, 0, stream>>>(k_pre, k_bf, KVc, 1);
  rope_relayout_kernel<<<(Bc * KVc * Sc * HDc) / 256, blk256, 0, stream>>>(v_pre, v_bf, KVc, 0);

  sumqkb_kernel<<<Bc * HDc, blk256, 0, stream>>>(qkb_bf, qsum);
  topk_kernel<<<Bc, blk256, 0, stream>>>(qsum, kb_keys, tidx);
  gather_kernel<<<(Bc * TOPK * HDc) / 256, blk256, 0, stream>>>(kb_keys, kb_values, tidx, kbk, kbv);

  attn_kernel<<<Bc * Hc * (Sc / 64), blk128, 0, stream>>>(q_bf, qkb_bf, k_bf, v_bf, kbk, kbv, score_shift, aout);

  gemm_xwt_kernel<true><<<dim3(HIDc / 128, M / 128), blk256, 0, stream>>>(aout, Wo, out, M, HIDc, HIDc);
}
